// TrajPreLocalAttnLong_31241592111496
// MI455X (gfx1250) — matrix operand order repaired and checked by reading, not compiled, not run on hardware
//
#include <hip/hip_runtime.h>
#include <hip/hip_bf16.h>
#include <math.h>

// ---------------- problem constants (from reference) ----------------
#define S_TOT 5120
#define T_LEN 1024
#define S_ENC 4096   // S - T
#define HDIM  512
#define DL 512
#define DT 32
#define DC 64
#define DD 608       // DL+DT+DC
#define G4 2048      // 4*H
#define NLOC 50000
#define DU 64
#define OUTD 1088    // 2H + DU

typedef __attribute__((ext_vector_type(16))) __bf16 v16bf;
typedef __attribute__((ext_vector_type(8)))  float  v8f;

// ==================================================================
// Generic bf16-WMMA GEMM:  C[M,N] (f32) = A[M,K] (f32) x B (f32)
//   BT=true : B stored [N,K] row-major  -> computes A @ B^T
//   BT=false: B stored [K,N] row-major  -> computes A @ B
// f32 operands converted to bf16 in registers; accumulate f32 via
// v_wmma_f32_16x16x32_bf16. Each wave owns a 32x32 register tile
// (2 A-tiles x 2 B-tiles -> 4 WMMA per k-step, operands reused).
// Tile linearization is M-fastest so concurrent waves share the same
// B strip: the streamed operand (weights) is read from HBM ~once and
// reused across all M out of L2; A is small and L2-resident.
// Requirements: K % 32 == 0, M % 32 == 0 (true for all calls here).
// ==================================================================

__device__ __forceinline__ v16bf load_a_tile(const float* __restrict__ Arow, int kk)
{
    // A 16x32 bf16 operand, contiguous k order used here: lane L: M = L&15, g = L>>4;
    // element e -> K = kk + 16g + e
    float4 a0 = *(const float4*)(Arow + kk);
    float4 a1 = *(const float4*)(Arow + kk + 4);
    float4 a2 = *(const float4*)(Arow + kk + 8);
    float4 a3 = *(const float4*)(Arow + kk + 12);
    v16bf v;
    v[0]=(__bf16)a0.x;  v[1]=(__bf16)a0.y;  v[2]=(__bf16)a0.z;  v[3]=(__bf16)a0.w;
    v[4]=(__bf16)a1.x;  v[5]=(__bf16)a1.y;  v[6]=(__bf16)a1.z;  v[7]=(__bf16)a1.w;
    v[8]=(__bf16)a2.x;  v[9]=(__bf16)a2.y;  v[10]=(__bf16)a2.z; v[11]=(__bf16)a2.w;
    v[12]=(__bf16)a3.x; v[13]=(__bf16)a3.y; v[14]=(__bf16)a3.z; v[15]=(__bf16)a3.w;
    return v;
}

template<bool BT>
__device__ __forceinline__ v16bf load_b_tile(const float* __restrict__ B,
                                             int n, int kbase, int ldb)
{
    // B 32x16 bf16 layout: lane L: N = L&15, g = L>>4; element e -> K = kbase + e (kbase = kk + 16g)
    v16bf v;
    if (BT) {
        const float* p = B + (size_t)n * ldb + kbase;   // [N,K] storage: contiguous K
        float4 b0 = *(const float4*)(p);
        float4 b1 = *(const float4*)(p + 4);
        float4 b2 = *(const float4*)(p + 8);
        float4 b3 = *(const float4*)(p + 12);
        v[0]=(__bf16)b0.x;  v[1]=(__bf16)b0.y;  v[2]=(__bf16)b0.z;  v[3]=(__bf16)b0.w;
        v[4]=(__bf16)b1.x;  v[5]=(__bf16)b1.y;  v[6]=(__bf16)b1.z;  v[7]=(__bf16)b1.w;
        v[8]=(__bf16)b2.x;  v[9]=(__bf16)b2.y;  v[10]=(__bf16)b2.z; v[11]=(__bf16)b2.w;
        v[12]=(__bf16)b3.x; v[13]=(__bf16)b3.y; v[14]=(__bf16)b3.z; v[15]=(__bf16)b3.w;
    } else {
        const float* p = B + (size_t)kbase * ldb + n;   // [K,N] storage: stride ldb
        #pragma unroll
        for (int e = 0; e < 16; ++e)
            v[e] = (__bf16)p[(size_t)e * ldb];
    }
    return v;
}

template<bool BT>
__global__ void gemm_wmma_bf16(const float* __restrict__ A,
                               const float* __restrict__ B,
                               float* __restrict__ C,
                               int M, int N, int K,
                               int lda, int ldb, int ldc)
{
    const int lane = threadIdx.x & 31;
    const int wave = threadIdx.x >> 5;
    const int tiles_n = (N + 31) >> 5;        // strips of 32 columns (2 tiles)
    const int tiles_m = M >> 5;               // strips of 32 rows    (2 tiles)
    const int gt = blockIdx.x * (blockDim.x >> 5) + wave;
    if (gt >= tiles_m * tiles_n) return;      // wave-uniform

    const int tm = gt % tiles_m;              // M-fastest: concurrent waves share B strip
    const int tn = gt / tiles_m;
    const int row0 = tm << 5;
    const int col0 = tn << 5;
    const int g  = lane >> 4;
    const int mr = lane & 15;
    const bool have2 = (col0 + 32) <= N;      // wave-uniform guard (N=50000 tail)

    v8f acc00 = {0.f,0.f,0.f,0.f,0.f,0.f,0.f,0.f};   // (M-tile 0, N-tile 0)
    v8f acc01 = acc00;                                // (M-tile 0, N-tile 1)
    v8f acc10 = acc00;                                // (M-tile 1, N-tile 0)
    v8f acc11 = acc00;                                // (M-tile 1, N-tile 1)

    const float* Arow0 = A + (size_t)(row0 + mr) * lda + 16 * g;
    const float* Arow1 = A + (size_t)(row0 + 16 + mr) * lda + 16 * g;
    const int kb0 = 16 * g;

    for (int kk = 0; kk < K; kk += 32) {
        if (kk + 32 < K) {
            __builtin_prefetch(Arow0 + kk + 32, 0, 1);   // global_prefetch_b8
            if (BT) __builtin_prefetch(B + (size_t)(col0 + mr) * ldb + kk + 32, 0, 1);
        }
        v16bf va0 = load_a_tile(Arow0, kk);
        v16bf va1 = load_a_tile(Arow1, kk);
        v16bf vb0 = load_b_tile<BT>(B, col0 + mr, kk + kb0, ldb);
        acc00 = __builtin_amdgcn_wmma_f32_16x16x32_bf16(false, va0, false, vb0,
                                                        (short)0, acc00, false, false);
        acc10 = __builtin_amdgcn_wmma_f32_16x16x32_bf16(false, va1, false, vb0,
                                                        (short)0, acc10, false, false);
        if (have2) {
            v16bf vb1 = load_b_tile<BT>(B, col0 + 16 + mr, kk + kb0, ldb);
            acc01 = __builtin_amdgcn_wmma_f32_16x16x32_bf16(false, va0, false, vb1,
                                                            (short)0, acc01, false, false);
            acc11 = __builtin_amdgcn_wmma_f32_16x16x32_bf16(false, va1, false, vb1,
                                                            (short)0, acc11, false, false);
        }
    }

    // C/D layout: element r -> row = rowbase + r + 8g, col = colbase + (lane&15)
    #pragma unroll
    for (int r = 0; r < 8; ++r) {
        const size_t ro0 = (size_t)(row0 + r + 8 * g) * ldc;
        const size_t ro1 = (size_t)(row0 + 16 + r + 8 * g) * ldc;
        C[ro0 + col0 + mr] = acc00[r];
        C[ro1 + col0 + mr] = acc10[r];
        if (have2) {
            C[ro0 + col0 + 16 + mr] = acc01[r];
            C[ro1 + col0 + 16 + mr] = acc11[r];
        }
    }
}

// ==================================================================
// Embedding gather:  x[s, 0:512|512:544|544:608] = emb_*[idx[s]]
// ==================================================================
__global__ void gather_embed(const int* __restrict__ loc, const int* __restrict__ tim,
                             const int* __restrict__ clu,
                             const float* __restrict__ el, const float* __restrict__ et,
                             const float* __restrict__ ec, float* __restrict__ x)
{
    int idx = blockIdx.x * blockDim.x + threadIdx.x;
    const int total = S_TOT * DD;
    if (idx >= total) return;
    int s = idx / DD, d = idx - s * DD;
    float v;
    if (d < DL)            v = el[(size_t)loc[s] * DL + d];
    else if (d < DL + DT)  v = et[(size_t)tim[s] * DT + (d - DL)];
    else                   v = ec[(size_t)clu[s] * DC + (d - DL - DT)];
    x[idx] = v;
}

// ==================================================================
// Persistent LSTM recurrence. gates_pre = x@Wih^T precomputed.
// grid.x = 2: block 0 = encoder (4096 steps), block 1 = decoder (1024).
// 1024 threads: thread t computes gates j=t and j=t+1024 each step;
// threads 0..511 apply the cell. h and gates live in LDS; Whh (4 MB)
// streams from L2 each step.
// ==================================================================
__global__ void __launch_bounds__(1024)
lstm_recurrent(const float* __restrict__ gates_pre,
               const float* __restrict__ encWhh, const float* __restrict__ decWhh,
               const float* __restrict__ enc_bih, const float* __restrict__ enc_bhh,
               const float* __restrict__ dec_bih, const float* __restrict__ dec_bhh,
               float* __restrict__ Hh, float* __restrict__ Hs)
{
    __shared__ float h[HDIM];
    __shared__ float gsh[G4];

    const int which = blockIdx.x;                      // 0 = enc, 1 = dec
    const float* Whh = which ? decWhh : encWhh;
    const float* bih = which ? dec_bih : enc_bih;
    const float* bhh = which ? dec_bhh : enc_bhh;
    const float* gp  = gates_pre + (which ? (size_t)S_ENC * G4 : 0);
    float* out       = which ? Hs : Hh;
    const int steps  = which ? T_LEN : S_ENC;

    const int tid = threadIdx.x;
    if (tid < HDIM) h[tid] = 0.f;
    float c = 0.f;
    const float b0 = bih[tid] + bhh[tid];
    const float b1 = bih[tid + 1024] + bhh[tid + 1024];
    const float4* w0 = (const float4*)(Whh + (size_t)tid * HDIM);
    const float4* w1 = (const float4*)(Whh + (size_t)(tid + 1024) * HDIM);
    __syncthreads();

    for (int t = 0; t < steps; ++t) {
        float s0 = gp[(size_t)t * G4 + tid] + b0;
        float s1 = gp[(size_t)t * G4 + tid + 1024] + b1;
        const float4* hv = (const float4*)h;
        #pragma unroll 4
        for (int k = 0; k < HDIM / 4; ++k) {
            float4 hk = hv[k];
            float4 a = w0[k], b = w1[k];
            s0 = fmaf(a.x, hk.x, fmaf(a.y, hk.y, fmaf(a.z, hk.z, fmaf(a.w, hk.w, s0))));
            s1 = fmaf(b.x, hk.x, fmaf(b.y, hk.y, fmaf(b.z, hk.z, fmaf(b.w, hk.w, s1))));
        }
        gsh[tid] = s0;
        gsh[tid + 1024] = s1;
        __syncthreads();
        if (tid < HDIM) {
            float gi = gsh[tid];
            float gf = gsh[HDIM + tid];
            float gg = gsh[2 * HDIM + tid];
            float go = gsh[3 * HDIM + tid];
            float si = 1.f / (1.f + __expf(-gi));
            float sf = 1.f / (1.f + __expf(-gf));
            float so = 1.f / (1.f + __expf(-go));
            c = sf * c + si * tanhf(gg);
            float hn = so * tanhf(c);
            h[tid] = hn;
            out[(size_t)t * HDIM + tid] = hn;
        }
        __syncthreads();
    }
}

// ==================================================================
// Row softmax in place (scores [T, S_ENC])
// ==================================================================
__global__ void softmax_rows(float* __restrict__ sc, int cols)
{
    __shared__ float red[256];
    float* r = sc + (size_t)blockIdx.x * cols;
    const int tid = threadIdx.x, bs = blockDim.x;
    float m = -1e30f;
    for (int c0 = tid; c0 < cols; c0 += bs) m = fmaxf(m, r[c0]);
    red[tid] = m; __syncthreads();
    for (int s = bs >> 1; s > 0; s >>= 1) { if (tid < s) red[tid] = fmaxf(red[tid], red[tid + s]); __syncthreads(); }
    m = red[0]; __syncthreads();
    float sum = 0.f;
    for (int c0 = tid; c0 < cols; c0 += bs) { float e = __expf(r[c0] - m); r[c0] = e; sum += e; }
    red[tid] = sum; __syncthreads();
    for (int s = bs >> 1; s > 0; s >>= 1) { if (tid < s) red[tid] += red[tid + s]; __syncthreads(); }
    const float inv = 1.f / red[0];
    for (int c0 = tid; c0 < cols; c0 += bs) r[c0] *= inv;
}

// ==================================================================
// Concat [hidden_state | context | uid_emb]
// ==================================================================
__global__ void concat_out(const float* __restrict__ Hs, const float* __restrict__ ctx,
                           const float* __restrict__ emb_uid, const int* __restrict__ uid,
                           float* __restrict__ outcat)
{
    int idx = blockIdx.x * blockDim.x + threadIdx.x;
    const int total = T_LEN * OUTD;
    if (idx >= total) return;
    int t = idx / OUTD, d = idx - t * OUTD;
    float v;
    if (d < HDIM)          v = Hs[(size_t)t * HDIM + d];
    else if (d < 2 * HDIM) v = ctx[(size_t)t * HDIM + (d - HDIM)];
    else                   v = emb_uid[(size_t)uid[0] * DU + (d - 2 * HDIM)];
    outcat[idx] = v;
}

// ==================================================================
// Row log-softmax with fused bias add:  y[t,:] = (y+fcb) - logsumexp
// ==================================================================
__global__ void logsoftmax_bias(float* __restrict__ y, const float* __restrict__ fcb, int cols)
{
    __shared__ float red[512];
    float* r = y + (size_t)blockIdx.x * cols;
    const int tid = threadIdx.x, bs = blockDim.x;
    float m = -1e30f;
    for (int c0 = tid; c0 < cols; c0 += bs) m = fmaxf(m, r[c0] + fcb[c0]);
    red[tid] = m; __syncthreads();
    for (int s = bs >> 1; s > 0; s >>= 1) { if (tid < s) red[tid] = fmaxf(red[tid], red[tid + s]); __syncthreads(); }
    m = red[0]; __syncthreads();
    float sum = 0.f;
    for (int c0 = tid; c0 < cols; c0 += bs) sum += __expf(r[c0] + fcb[c0] - m);
    red[tid] = sum; __syncthreads();
    for (int s = bs >> 1; s > 0; s >>= 1) { if (tid < s) red[tid] += red[tid + s]; __syncthreads(); }
    const float lse = m + __logf(red[0]);
    for (int c0 = tid; c0 < cols; c0 += bs) r[c0] = r[c0] + fcb[c0] - lse;
}

// ==================================================================
// Host launcher
// ==================================================================
template<bool BT>
static inline void launch_gemm(const float* A, const float* B, float* C,
                               int M, int N, int K, int lda, int ldb, int ldc,
                               hipStream_t stream)
{
    int tiles_n = (N + 31) / 32;
    int tiles_m = M / 32;                 // all M here are multiples of 32
    int ntiles  = tiles_m * tiles_n;
    int wpb     = 256 / 32;
    int grid    = (ntiles + wpb - 1) / wpb;
    gemm_wmma_bf16<BT><<<grid, 256, 0, stream>>>(A, B, C, M, N, K, lda, ldb, ldc);
}

extern "C" void kernel_launch(void* const* d_in, const int* in_sizes, int n_in,
                              void* d_out, int out_size, void* d_ws, size_t ws_size,
                              hipStream_t stream)
{
    const int*   loc      = (const int*)d_in[0];
    const int*   tim      = (const int*)d_in[1];
    const int*   clu      = (const int*)d_in[2];
    const int*   uid      = (const int*)d_in[3];
    /* d_in[4] = target_len (compile-time T_LEN) */
    const float* emb_loc  = (const float*)d_in[5];
    const float* emb_tim  = (const float*)d_in[6];
    const float* emb_clu  = (const float*)d_in[7];
    const float* emb_uid  = (const float*)d_in[8];
    const float* enc_Wih  = (const float*)d_in[9];
    const float* enc_Whh  = (const float*)d_in[10];
    const float* enc_bih  = (const float*)d_in[11];
    const float* enc_bhh  = (const float*)d_in[12];
    const float* dec_Wih  = (const float*)d_in[13];
    const float* dec_Whh  = (const float*)d_in[14];
    const float* dec_bih  = (const float*)d_in[15];
    const float* dec_bhh  = (const float*)d_in[16];
    const float* fcW      = (const float*)d_in[17];
    const float* fcb      = (const float*)d_in[18];
    float* y = (float*)d_out;                        // [T, NLOC]

    // workspace carve-out (~88 MB f32)
    float* ws     = (float*)d_ws;
    float* x      = ws;                               // [S, 608]
    float* gates  = x      + (size_t)S_TOT * DD;      // [S, 2048]
    float* Hh     = gates  + (size_t)S_TOT * G4;      // [4096, 512]
    float* Hs     = Hh     + (size_t)S_ENC * HDIM;    // [1024, 512]
    float* scores = Hs     + (size_t)T_LEN * HDIM;    // [1024, 4096] (softmax in place)
    float* ctx    = scores + (size_t)T_LEN * S_ENC;   // [1024, 512]
    float* outcat = ctx    + (size_t)T_LEN * HDIM;    // [1024, 1088]

    // 1) gather embeddings -> x
    {
        int total = S_TOT * DD;
        gather_embed<<<(total + 255) / 256, 256, 0, stream>>>(loc, tim, clu,
                                                              emb_loc, emb_tim, emb_clu, x);
    }

    // 2) input projections (non-recurrent part of both LSTMs), WMMA bf16
    launch_gemm<true>(x, enc_Wih, gates,
                      S_ENC, G4, DD, DD, DD, G4, stream);
    launch_gemm<true>(x + (size_t)S_ENC * DD, dec_Wih, gates + (size_t)S_ENC * G4,
                      T_LEN, G4, DD, DD, DD, G4, stream);

    // 3) both recurrences concurrently (encoder & decoder are independent)
    lstm_recurrent<<<2, 1024, 0, stream>>>(gates, enc_Whh, dec_Whh,
                                           enc_bih, enc_bhh, dec_bih, dec_bhh, Hh, Hs);

    // 4) attention scores = Hs @ Hh^T   (WMMA)
    launch_gemm<true>(Hs, Hh, scores, T_LEN, S_ENC, HDIM, HDIM, HDIM, S_ENC, stream);

    // 5) softmax rows (in place -> attention probs)
    softmax_rows<<<T_LEN, 256, 0, stream>>>(scores, S_ENC);

    // 6) context = P @ Hh   (WMMA, B stored [K,N])
    launch_gemm<false>(scores, Hh, ctx, T_LEN, HDIM, S_ENC, S_ENC, HDIM, HDIM, stream);

    // 7) concat [Hs | ctx | uid_emb]
    {
        int total = T_LEN * OUTD;
        concat_out<<<(total + 255) / 256, 256, 0, stream>>>(Hs, ctx, emb_uid, uid, outcat);
    }

    // 8) final FC: y = outcat @ fcW^T  (111 GFLOP, WMMA bf16, fcW converted in-register)
    launch_gemm<true>(outcat, fcW, y, T_LEN, NLOC, OUTD, OUTD, OUTD, NLOC, stream);

    // 9) fused bias + log_softmax per row
    logsoftmax_bias<<<T_LEN, 512, 0, stream>>>(y, fcb, NLOC);
}
